// BVPModel_76871324664177
// MI455X (gfx1250) — compile-verified, matrix operand order checked by reading, not run on hardware
//
#include <hip/hip_runtime.h>

// ---------------------------------------------------------------------------
// BVP PINN residual: fused 2nd-order forward-jet MLP on CDNA5 WMMA.
//   7 jet channels (value, t_x,t_y,t_z, s_x,s_y,s_z) share each weight matrix;
//   hidden layers are [16 x 256] @ [256 x 256] GEMMs on v_wmma_f32_16x16x32_f16.
//   Weight K-slices (16 KB) are DMA'd to LDS once per workgroup by the Tensor
//   Data Mover (double-buffered, TENSORcnt), cutting L2 weight re-reads 7x and
//   decoupling WMMA issue from memory latency. TDM LDS padding (+4 dwords per
//   16-dword row) and 264-half activation rows kill LDS bank conflicts.
// ---------------------------------------------------------------------------

typedef __attribute__((ext_vector_type(16))) _Float16 v16h;
typedef __attribute__((ext_vector_type(8)))  float    v8f;
typedef unsigned int u32x4 __attribute__((ext_vector_type(4)));
typedef int          i32x8 __attribute__((ext_vector_type(8)));
typedef int          i32x4 __attribute__((ext_vector_type(4)));

#define NPTS   131072
#define HID    256
#define MTILE  16            // points per workgroup
#define NWAVES 7             // one wave per jet channel
#define ASTR   264           // padded activation row stride (halves); 132 dw % 64 = 4
#define WSTR   40            // TDM-padded W-slice row stride (halves); 20 dw % 64 = 20
#define SLICE_ELE (HID * WSTR)
#define PI_F   3.14159265358979f

union Frag16 { v16h h; uint4 q[2]; };

// Convert W2 / W3 (f32, 256x256 row-major [out][in]) to f16 in workspace.
__global__ void bvp_convert_w(const float* __restrict__ W2,
                              const float* __restrict__ W3,
                              _Float16* __restrict__ W2h,
                              _Float16* __restrict__ W3h) {
    int i = blockIdx.x * blockDim.x + threadIdx.x;
    if (i < HID * HID) {
        W2h[i] = (_Float16)W2[i];
        W3h[i] = (_Float16)W3[i];
    }
}

__device__ __forceinline__ unsigned ldsoff(const void* p) {
    // generic pointers to LDS carry the wave-relative byte offset in [31:0]
    return (unsigned)(unsigned long long)p;
}

// TDM: load one [256 rows x 32 halves] K-slice of W (row-major, stride 256)
// into LDS with +4-dword padding after every 16-dword row (row stride 80 B).
__device__ __forceinline__ void tdm_load_slice(const _Float16* gsrc, unsigned lds_byte) {
    unsigned long long ga = (unsigned long long)(const void*)gsrc;
    u32x4 g0;
    g0[0] = 1u;                                            // count=1, user desc
    g0[1] = lds_byte;                                      // lds_addr
    g0[2] = (unsigned)ga;                                  // global_addr[31:0]
    g0[3] = (unsigned)((ga >> 32) & 0x01FFFFFFu) | (2u << 30); // addr[56:32]|type=2
    i32x8 g1;
    g1[0] = (int)0x06D10000u;   // data_size=2B | pad_en | pad_interval=16dw | pad_amt=4dw
    g1[1] = (int)(256u << 16);  // tensor_dim0 = 256
    g1[2] = (int)(256u << 16);  // tensor_dim1 = 256
    g1[3] = (int)(32u  << 16);  // tile_dim0   = 32
    g1[4] = (int)256;           // tile_dim1   = 256
    g1[5] = (int)256;           // tensor_dim0_stride = 256
    g1[6] = 0;
    g1[7] = 0;
    i32x4 z4 = {0, 0, 0, 0};
    i32x8 z8 = {0, 0, 0, 0, 0, 0, 0, 0};
    __builtin_amdgcn_tensor_load_to_lds(g0, g1, z4, z4, z8, 0);
}

// One channel's [16 x 256] @ W^T[256 x 256] -> acc[16 n-tiles][v8f], with the
// weight K-slices streamed into LDS by the TDM (double-buffered).
__device__ __forceinline__ void gemm_layer(const _Float16* __restrict__ Ash_w,
                                           const _Float16* __restrict__ Wh,
                                           _Float16 (*Wslice)[SLICE_ELE],
                                           v8f acc[16], int lane, int w) {
    if (w == 0) {
        tdm_load_slice(Wh, ldsoff(&Wslice[0][0]));
        __builtin_amdgcn_s_wait_tensorcnt(0);
    }
    __syncthreads();
    const int mn  = lane & 15;
    const int sel = lane >> 4;
    for (int kt = 0; kt < 8; ++kt) {
        if (w == 0 && kt < 7)   // overlap DMA of next slice with this slice's WMMAs
            tdm_load_slice(Wh + (kt + 1) * 32, ldsoff(&Wslice[(kt + 1) & 1][0]));
        Frag16 a;
        a.q[0] = *(const uint4*)(Ash_w + mn * ASTR + kt * 32 + sel * 8);
        a.q[1] = *(const uint4*)(Ash_w + mn * ASTR + kt * 32 + 16 + sel * 8);
        const _Float16* ws = &Wslice[kt & 1][0];
#pragma unroll
        for (int nt = 0; nt < 16; ++nt) {
            Frag16 b;
            const _Float16* wrow = ws + (nt * 16 + mn) * WSTR + sel * 8;
            b.q[0] = *(const uint4*)(wrow);
            b.q[1] = *(const uint4*)(wrow + 16);
            acc[nt] = __builtin_amdgcn_wmma_f32_16x16x32_f16(
                false, a.h, false, b.h, (short)0, acc[nt], false, false);
        }
        if (w == 0 && kt < 7) __builtin_amdgcn_s_wait_tensorcnt(0);
        __syncthreads();
    }
}

__device__ __forceinline__ void reduce_store(float pr[8], float pim[8],
                                             int slot, int lane,
                                             float (*pres)[2][MTILE]) {
#pragma unroll
    for (int r = 0; r < 8; ++r) {
#pragma unroll
        for (int mask = 1; mask < 16; mask <<= 1) {
            pr[r]  += __shfl_xor(pr[r],  mask, 32);
            pim[r] += __shfl_xor(pim[r], mask, 32);
        }
    }
    if ((lane & 15) == 0) {
        const int mb = (lane >> 4) * 8;
#pragma unroll
        for (int r = 0; r < 8; ++r) {
            pres[slot][0][mb + r] = pr[r];
            pres[slot][1][mb + r] = pim[r];
        }
    }
}

__global__ __launch_bounds__(NWAVES * 32, 1)
void bvp_residual_kernel(const float* __restrict__ x, const float* __restrict__ y,
                         const float* __restrict__ z, const float* __restrict__ f,
                         const float* __restrict__ W1, const float* __restrict__ b1,
                         const float* __restrict__ b2, const float* __restrict__ b3,
                         const float* __restrict__ W4, const float* __restrict__ b4,
                         const _Float16* __restrict__ W2h,
                         const _Float16* __restrict__ W3h,
                         float* __restrict__ out) {
    __shared__ _Float16 Ash[NWAVES][MTILE * ASTR];     // 57.75 KB (padded rows)
    __shared__ float    a_val[MTILE * ASTR];           // 16.5 KB value activation
    __shared__ _Float16 utb[3][MTILE * ASTR];          // 24.75 KB tangent pre-acts
    __shared__ _Float16 Wslice[2][SLICE_ELE];          // 40 KB TDM double buffer
    __shared__ float    pres[4][2][MTILE];

    const int tid  = threadIdx.x;
    const int w    = tid >> 5;      // wave = channel: 0 val, 1-3 tan, 4-6 snd
    const int lane = tid & 31;
    const int p0   = blockIdx.x * MTILE;
    const int ncol = lane & 15;
    const int mr0  = (lane >> 4) * 8;

    // ---------------- layer 1 (4 -> 256), VALU, no cross-wave deps ---------
    {
        const int m  = lane >> 1;
        const int jb = (lane & 1) << 7;
        const float xv = x[p0 + m], yv = y[p0 + m];
        const float zv = z[p0 + m], fv = f[p0 + m];
        const int dir = (w == 0) ? 0 : ((w <= 3) ? (w - 1) : (w - 4));
        for (int i = 0; i < 128; ++i) {
            const int j = jb + i;
            float u = W1[j * 4 + 0] * xv + W1[j * 4 + 1] * yv +
                      W1[j * 4 + 2] * zv + W1[j * 4 + 3] * fv + b1[j];
            float a = tanhf(u);
            float g = 1.f - a * a;
            float d = W1[j * 4 + dir];
            float o = (w == 0) ? a : ((w <= 3) ? g * d : -2.f * a * g * d * d);
            Ash[w][m * ASTR + j] = (_Float16)o;
        }
    }

    // ---------------- layer 2 (256 -> 256) --------------------------------
    v8f acc[16];
#pragma unroll
    for (int nt = 0; nt < 16; ++nt) { v8f zz = {}; acc[nt] = zz; }
    gemm_layer(Ash[w], W2h, Wslice, acc, lane, w);

    if (w == 0) {                                  // value: a = tanh(u + b2)
#pragma unroll
        for (int nt = 0; nt < 16; ++nt) {
            const int n = nt * 16 + ncol;
            const float bn = b2[n];
#pragma unroll
            for (int r = 0; r < 8; ++r) {
                float a = tanhf(acc[nt][r] + bn);
                a_val[(mr0 + r) * ASTR + n] = a;
                Ash[0][(mr0 + r) * ASTR + n] = (_Float16)a;
            }
        }
    } else if (w <= 3) {                           // tangent: publish u_t
        const int d = w - 1;
#pragma unroll
        for (int nt = 0; nt < 16; ++nt) {
            const int n = nt * 16 + ncol;
#pragma unroll
            for (int r = 0; r < 8; ++r)
                utb[d][(mr0 + r) * ASTR + n] = (_Float16)acc[nt][r];
        }
    }
    __syncthreads();
    if (w >= 1 && w <= 3) {                        // t+ = a' * u_t
#pragma unroll
        for (int nt = 0; nt < 16; ++nt) {
            const int n = nt * 16 + ncol;
#pragma unroll
            for (int r = 0; r < 8; ++r) {
                float a = a_val[(mr0 + r) * ASTR + n];
                Ash[w][(mr0 + r) * ASTR + n] =
                    (_Float16)((1.f - a * a) * acc[nt][r]);
            }
        }
    } else if (w >= 4) {                           // s+ = a'' u_t^2 + a' u_tt
        const int d = w - 4;
#pragma unroll
        for (int nt = 0; nt < 16; ++nt) {
            const int n = nt * 16 + ncol;
#pragma unroll
            for (int r = 0; r < 8; ++r) {
                float a = a_val[(mr0 + r) * ASTR + n];
                float g = 1.f - a * a;
                float t = (float)utb[d][(mr0 + r) * ASTR + n];
                Ash[w][(mr0 + r) * ASTR + n] =
                    (_Float16)(-2.f * a * g * t * t + g * acc[nt][r]);
            }
        }
    }
    __syncthreads();

    // ---------------- layer 3 (256 -> 256) --------------------------------
#pragma unroll
    for (int nt = 0; nt < 16; ++nt) { v8f zz = {}; acc[nt] = zz; }
    gemm_layer(Ash[w], W3h, Wslice, acc, lane, w);

    // Fused layer 4 (2 x 256 dots) on the jet outputs we still need.
    if (w == 0) {                                  // psi = W4 a + b4
        float pr[8] = {}, pim[8] = {};
#pragma unroll
        for (int nt = 0; nt < 16; ++nt) {
            const int n = nt * 16 + ncol;
            const float bn = b3[n], w4r = W4[n], w4i = W4[HID + n];
#pragma unroll
            for (int r = 0; r < 8; ++r) {
                float a = tanhf(acc[nt][r] + bn);
                a_val[(mr0 + r) * ASTR + n] = a;
                pr[r] += w4r * a;
                pim[r] += w4i * a;
            }
        }
        reduce_store(pr, pim, 0, lane, pres);
    } else if (w <= 3) {
        const int d = w - 1;
#pragma unroll
        for (int nt = 0; nt < 16; ++nt) {
            const int n = nt * 16 + ncol;
#pragma unroll
            for (int r = 0; r < 8; ++r)
                utb[d][(mr0 + r) * ASTR + n] = (_Float16)acc[nt][r];
        }
    }
    __syncthreads();
    if (w >= 4) {                                  // p_dd = W4 s
        const int d = w - 4;
        float pr[8] = {}, pim[8] = {};
#pragma unroll
        for (int nt = 0; nt < 16; ++nt) {
            const int n = nt * 16 + ncol;
            const float w4r = W4[n], w4i = W4[HID + n];
#pragma unroll
            for (int r = 0; r < 8; ++r) {
                float a = a_val[(mr0 + r) * ASTR + n];
                float g = 1.f - a * a;
                float t = (float)utb[d][(mr0 + r) * ASTR + n];
                float s = -2.f * a * g * t * t + g * acc[nt][r];
                pr[r] += w4r * s;
                pim[r] += w4i * s;
            }
        }
        reduce_store(pr, pim, 1 + d, lane, pres);
    }
    __syncthreads();

    // ---------------- residual combine & store ----------------------------
    if (w == 0 && lane < MTILE) {
        const int p = p0 + lane;
        const float XC = 0.7f, YC = 0.5f, ZC = 0.6f;
        const float AC = 2.0f, A0 = 0.1f, BC = 1.5f, B0 = -0.05f;
        float k = 2.f * PI_F * (f[p] * 500.f + 100.f) / 343.f;
        float kk2 = XC * YC * ZC * k; kk2 *= kk2;
        float cx = (YC * ZC) * (YC * ZC);
        float cy = (XC * ZC) * (XC * ZC);
        float cz = (XC * YC) * (XC * YC);
        float pr = pres[0][0][lane] + b4[0];
        float pi = pres[0][1][lane] + b4[1];
        float rre = AC * (cx * pres[1][0][lane] + cy * pres[2][0][lane] +
                          cz * pres[3][0][lane]) + kk2 * (pr * AC + A0);
        float rim = BC * (cx * pres[1][1][lane] + cy * pres[2][1][lane] +
                          cz * pres[3][1][lane]) + kk2 * (pi * BC + B0);
        out[p]        = rre;
        out[NPTS + p] = rim;
    }
}

extern "C" void kernel_launch(void* const* d_in, const int* in_sizes, int n_in,
                              void* d_out, int out_size, void* d_ws, size_t ws_size,
                              hipStream_t stream) {
    const float* x  = (const float*)d_in[0];
    const float* y  = (const float*)d_in[1];
    const float* z  = (const float*)d_in[2];
    const float* f  = (const float*)d_in[3];
    const float* W1 = (const float*)d_in[4];
    const float* b1 = (const float*)d_in[5];
    const float* W2 = (const float*)d_in[6];
    const float* b2 = (const float*)d_in[7];
    const float* W3 = (const float*)d_in[8];
    const float* b3 = (const float*)d_in[9];
    const float* W4 = (const float*)d_in[10];
    const float* b4 = (const float*)d_in[11];
    float* out = (float*)d_out;

    _Float16* W2h = (_Float16*)d_ws;
    _Float16* W3h = W2h + HID * HID;

    bvp_convert_w<<<(HID * HID + 255) / 256, 256, 0, stream>>>(W2, W3, W2h, W3h);
    bvp_residual_kernel<<<NPTS / MTILE, NWAVES * 32, 0, stream>>>(
        x, y, z, f, W1, b1, b2, b3, W4, b4, W2h, W3h, out);
}
